// EquiformerV2_19112604467722
// MI455X (gfx1250) — matrix operand order repaired and checked by reading, not compiled, not run on hardware
//
#include <hip/hip_runtime.h>
#include <hip/hip_bf16.h>

// ---------------- problem dims ----------------
constexpr int N  = 8192;
constexpr int E  = 65536;
constexpr int C  = 128;
constexpr int NB = 96;
constexpr int L  = 4;

typedef __attribute__((ext_vector_type(16))) __bf16 bf16x16;
typedef __attribute__((ext_vector_type(8)))  __bf16 bf16x8;
typedef __attribute__((ext_vector_type(8)))  float  f32x8;
typedef __attribute__((ext_vector_type(4)))  unsigned u32x4;
typedef __attribute__((ext_vector_type(4)))  int    i32x4;
typedef __attribute__((ext_vector_type(8)))  int    i32x8;

// ---------------- WMMA helpers (layouts per CDNA5 ISA 7.12.2) ----------------
__device__ __forceinline__ f32x8 wmma_bf16(bf16x16 a, bf16x16 b, f32x8 c) {
  return __builtin_amdgcn_wmma_f32_16x16x32_bf16(false, a, false, b, (short)0, c,
                                                 false, false);
}

// A fragment (16x32, M x K) from row-major f32, row stride ld (cvt path; agg only)
__device__ __forceinline__ bf16x16 a_frag_f32(const float* A, int ld, int lane) {
  const float* p = A + (size_t)(lane & 15) * ld + ((lane & 16) ? 8 : 0);
  bf16x16 a;
#pragma unroll
  for (int i = 0; i < 8; ++i) a[i] = (__bf16)p[i];
#pragma unroll
  for (int i = 0; i < 8; ++i) a[8 + i] = (__bf16)p[16 + i];
  return a;
}
// A fragment from row-major bf16: two contiguous 16B loads, no conversion
__device__ __forceinline__ bf16x16 a_frag_bf16(const __bf16* A, int ld, int lane) {
  const __bf16* p = A + (size_t)(lane & 15) * ld + ((lane & 16) ? 8 : 0);
  bf16x8 lo = *(const bf16x8*)(p);
  bf16x8 hi = *(const bf16x8*)(p + 16);
  bf16x16 a;
#pragma unroll
  for (int i = 0; i < 8; ++i) { a[i] = lo[i]; a[8 + i] = hi[i]; }
  return a;
}
// fragment from pre-swizzled fragment-major table (global or LDS): 32B contiguous
__device__ __forceinline__ bf16x16 frag_swz(const __bf16* Bs, int frag, int lane) {
  return *(const bf16x16*)(Bs + (((size_t)frag * 32 + lane) << 4));
}

__device__ __forceinline__ float silu_f(float x) { return x / (1.f + __expf(-x)); }

// order-preserving float<->uint encoding for atomic max
__device__ __forceinline__ unsigned fenc(float x) {
  unsigned u = __float_as_uint(x);
  return (u >> 31) ? ~u : (u | 0x80000000u);
}
__device__ __forceinline__ float fdec(unsigned u) {
  u = (u >> 31) ? (u & 0x7fffffffu) : ~u;
  return __uint_as_float(u);
}

// ---------------- TDM: 1-D bulk copy global -> LDS (n8 units of 8 bytes) ------
// D# per CDNA5 ISA: group0 {count=1, lds_addr,
// global_addr, type=2}; group1 {mask=0, data_size=3(8B), tensor_dim0=tile_dim0=n8}
__device__ __forceinline__ void tdm_load_1d(unsigned lds_off, const void* gptr,
                                            unsigned n8) {
  unsigned long long ga = (unsigned long long)gptr;
  u32x4 g0;
  g0[0] = 1u;                                              // count=1, user mode
  g0[1] = lds_off;                                         // lds_addr (bytes)
  g0[2] = (unsigned)(ga & 0xffffffffu);                    // global_addr[31:0]
  g0[3] = (unsigned)((ga >> 32) & 0x01ffffffu) | (2u << 30); // [56:32] | type=2
  i32x8 g1;
  g1[0] = (int)(3u << 16);                                 // data_size=3 (8B)
  g1[1] = (int)((n8 & 0xffffu) << 16);                     // tensor_dim0[15:0]
  g1[2] = (int)(n8 >> 16);                                 // tensor_dim0[31:16]
  g1[3] = (int)((n8 & 0xffffu) << 16);                     // tile_dim0
  g1[4] = 0;                                               // tile_dim1/2 unused
  g1[5] = (int)n8;                                         // tensor_dim0_stride
  g1[6] = 0; g1[7] = 0;
  i32x4 z4 = {0, 0, 0, 0};
#if __clang_major__ >= 23
  i32x8 z8 = {0, 0, 0, 0, 0, 0, 0, 0};
  __builtin_amdgcn_tensor_load_to_lds(g0, g1, z4, z4, z8, 0);
#else
  __builtin_amdgcn_tensor_load_to_lds(g0, g1, z4, z4, 0);
#endif
}

// ---------------- weight pre-swizzle: row-major f32 (K x Nn) -> fragment-major bf16
__global__ void k_prep_b(const float* __restrict__ W, __bf16* __restrict__ out,
                         int Nn, int NT, int total) {
  int d = blockIdx.x * 256 + threadIdx.x;
  if (d >= total) return;
  int i    = d & 15;
  int lane = (d >> 4) & 31;
  int f    = d >> 9;
  int nt = f % NT, kf = f / NT;
  int row = kf * 32 + ((lane & 16) ? 8 : 0) + (i & 7) + ((i & 8) ? 16 : 0);
  int col = nt * 16 + (lane & 15);
  out[d] = (__bf16)W[(size_t)row * Nn + col];
}

// ---------------- edge geometry: SH + RBF (rbf stored bf16) ----------------
__global__ void k_edge_geom(const float* __restrict__ pos, const int* __restrict__ src,
                            const int* __restrict__ dst, float* __restrict__ sh,
                            __bf16* __restrict__ rbf) {
  int e = blockIdx.x * blockDim.x + threadIdx.x;
  int sn = src[e], tn = dst[e];
  float vx = pos[3 * tn]     - pos[3 * sn];
  float vy = pos[3 * tn + 1] - pos[3 * sn + 1];
  float vz = pos[3 * tn + 2] - pos[3 * sn + 2];
  float d = sqrtf(vx * vx + vy * vy + vz * vz + 1e-12f);
  float inv = 1.f / d;
  float x = vx * inv, y = vy * inv, z = vz * inv;
  const float s3 = 1.7320508075688772f, s5 = 2.23606797749979f,
              s7 = 2.6457513110645907f, s15 = 3.872983346207417f,
              s42 = 6.48074069840786f,  s70 = 8.366600265340756f,
              s105 = 10.246950765959598f;
  float* shp = sh + (size_t)e * 16;
  shp[0] = 1.f;         shp[1] = s3 * x;      shp[2] = s3 * y;      shp[3] = s3 * z;
  shp[4] = s15 * x * y; shp[5] = s15 * y * z; shp[6] = 0.5f * s5 * (3.f * z * z - 1.f);
  shp[7] = s15 * x * z; shp[8] = 0.5f * s15 * (x * x - y * y);
  shp[9]  = 0.25f * s70 * y * (3.f * x * x - y * y);
  shp[10] = s105 * x * y * z;
  shp[11] = 0.25f * s42 * y * (5.f * z * z - 1.f);
  shp[12] = 0.5f  * s7  * z * (5.f * z * z - 3.f);
  shp[13] = 0.25f * s42 * x * (5.f * z * z - 1.f);
  shp[14] = 0.5f  * s105 * z * (x * x - y * y);
  shp[15] = 0.25f * s70 * x * (x * x - 3.f * y * y);
  __bf16* rp = rbf + (size_t)e * NB;
  const float invw = (float)NB / 10.f;
  const float step = 10.f / (float)(NB - 1);
#pragma unroll 4
  for (int j = 0; j < NB; ++j) {
    float t = (d - step * j) * invw;
    rp[j] = (__bf16)__expf(-0.5f * t * t);
  }
}

// ---------------- radial MLP: rad = silu(rbf@W1+b1)@W2  (E,96)->(E,64)->(E,128)
__global__ void __launch_bounds__(256) k_rad_mlp(const __bf16* __restrict__ rbf,
    const __bf16* __restrict__ w1s, const float* __restrict__ b1,
    const __bf16* __restrict__ w2s, float* __restrict__ rad) {
  __shared__ __align__(32) __bf16 hs[8][2 * 32 * 16];   // swizzled 16x64 per wave
  int tid = threadIdx.x, lane = tid & 31, wid = tid >> 5;
  int tile = blockIdx.x * 8 + wid;   // 16 edges per wave
  const __bf16* A = rbf + (size_t)tile * 16 * NB;
  f32x8 acc[4];
#pragma unroll
  for (int nt = 0; nt < 4; ++nt)
#pragma unroll
    for (int i = 0; i < 8; ++i) acc[nt][i] = 0.f;
#pragma unroll
  for (int k = 0; k < 3; ++k) {
    bf16x16 a = a_frag_bf16(A + k * 32, NB, lane);
#pragma unroll
    for (int nt = 0; nt < 4; ++nt)
      acc[nt] = wmma_bf16(a, frag_swz(w1s, k * 4 + nt, lane), acc[nt]);
  }
  int roff = (lane & 16) ? 8 : 0, col4 = lane & 15;
  int lhx = (col4 >> 3) << 4;
  __bf16* h = hs[wid];
#pragma unroll
  for (int nt = 0; nt < 4; ++nt) {
    float bb = b1[nt * 16 + col4];
    int kf = nt >> 1;
    int eb = (nt & 1) * 8 + (col4 & 7);
#pragma unroll
    for (int r = 0; r < 8; ++r)
      h[((kf * 32) + (r + roff) + lhx) * 16 + eb] = (__bf16)silu_f(acc[nt][r] + bb);
  }
  f32x8 acc2[8];
#pragma unroll
  for (int nt = 0; nt < 8; ++nt)
#pragma unroll
    for (int i = 0; i < 8; ++i) acc2[nt][i] = 0.f;
#pragma unroll
  for (int k = 0; k < 2; ++k) {
    bf16x16 a = frag_swz(h, k, lane);
#pragma unroll
    for (int nt = 0; nt < 8; ++nt)
      acc2[nt] = wmma_bf16(a, frag_swz(w2s, k * 8 + nt, lane), acc2[nt]);
  }
  float* out = rad + (size_t)tile * 16 * 128;
#pragma unroll
  for (int nt = 0; nt < 8; ++nt)
#pragma unroll
    for (int r = 0; r < 8; ++r)
      out[(r + roff) * 128 + nt * 16 + col4] = acc2[nt][r];
}

// ---------------- x init + message scatter ----------------
__global__ void k_init_x(const float* __restrict__ nf, float* __restrict__ x) {
  size_t i = (size_t)blockIdx.x * 256 + threadIdx.x;  // over N*2048
  int n = (int)(i >> 11);
  int r = (int)(i & 2047);
  x[i] = (r < C) ? nf[(size_t)n * C + r] : 0.f;
}

__global__ void k_msg_scatter(const float* __restrict__ sh, const float* __restrict__ rad,
                              const int* __restrict__ dst, float* __restrict__ x) {
  int e = blockIdx.x, c = threadIdx.x;   // 128 threads
  float rc = rad[(size_t)e * 128 + c] * 0.125f;
  float* xp = x + (size_t)dst[e] * 2048 + c;
  const float* shp = sh + (size_t)e * 16;
#pragma unroll
  for (int l = 0; l < 16; ++l) atomicAdd(xp + l * 128, shp[l] * rc);
}

// ---------------- per-degree RMS norm (writes bf16 xn) ----------------
__global__ void k_rmsnorm(const float* __restrict__ x, const float* __restrict__ gamma,
                          __bf16* __restrict__ xn) {
  int n = blockIdx.x, c = threadIdx.x;   // 128 threads
  const float* xp = x + (size_t)n * 2048 + c;
  float v[16];
#pragma unroll
  for (int l = 0; l < 16; ++l) v[l] = xp[l * 128];
  float s0 = v[0] * v[0];
  float s1 = v[1] * v[1] + v[2] * v[2] + v[3] * v[3];
  float s2 = v[4] * v[4] + v[5] * v[5] + v[6] * v[6] + v[7] * v[7] + v[8] * v[8];
  float s3 = 0.f;
#pragma unroll
  for (int l = 9; l < 16; ++l) s3 += v[l] * v[l];
  float inv[4] = { rsqrtf(s0 + 1e-6f), rsqrtf(s1 * (1.f / 3.f) + 1e-6f),
                   rsqrtf(s2 * 0.2f + 1e-6f), rsqrtf(s3 * (1.f / 7.f) + 1e-6f) };
  float g[4] = { gamma[c], gamma[128 + c], gamma[256 + c], gamma[384 + c] };
  __bf16* op = xn + (size_t)n * 2048 + c;
  const int deg[16] = {0,1,1,1,2,2,2,2,2,3,3,3,3,3,3,3};
#pragma unroll
  for (int l = 0; l < 16; ++l) op[l * 128] = (__bf16)(v[l] * inv[deg[l]] * g[deg[l]]);
}

// ---------------- erad = silu(rbf@Wr+b) (E,96)->(E,64), bf16 out -------------
__global__ void __launch_bounds__(256) k_erad(const __bf16* __restrict__ rbf,
    const __bf16* __restrict__ w1s, const float* __restrict__ b1,
    __bf16* __restrict__ erad) {
  int tid = threadIdx.x, lane = tid & 31, wid = tid >> 5;
  int tile = blockIdx.x * 8 + wid;
  const __bf16* A = rbf + (size_t)tile * 16 * NB;
  f32x8 acc[4];
#pragma unroll
  for (int nt = 0; nt < 4; ++nt)
#pragma unroll
    for (int i = 0; i < 8; ++i) acc[nt][i] = 0.f;
#pragma unroll
  for (int k = 0; k < 3; ++k) {
    bf16x16 a = a_frag_bf16(A + k * 32, NB, lane);
#pragma unroll
    for (int nt = 0; nt < 4; ++nt)
      acc[nt] = wmma_bf16(a, frag_swz(w1s, k * 4 + nt, lane), acc[nt]);
  }
  int roff = (lane & 16) ? 8 : 0, col4 = lane & 15;
  __bf16* out = erad + (size_t)tile * 16 * 64;
#pragma unroll
  for (int nt = 0; nt < 4; ++nt) {
    float bb = b1[nt * 16 + col4];
#pragma unroll
    for (int r = 0; r < 8; ++r)
      out[(r + roff) * 64 + nt * 16 + col4] = (__bf16)silu_f(acc[nt][r] + bb);
  }
}

// ---------------- attention logits: concat(320)@W1 -> silu -> @W2 -> (E,4)
__global__ void __launch_bounds__(128) k_alpha(const __bf16* __restrict__ xn,
    const int* __restrict__ src, const int* __restrict__ dst,
    const __bf16* __restrict__ erad, const __bf16* __restrict__ w1s,
    const float* __restrict__ b1, const float* __restrict__ w2,
    float* __restrict__ logits) {
  __shared__ float hs[4][16 * 64];
  int tid = threadIdx.x, lane = tid & 31, wid = tid >> 5;
  int tile = blockIdx.x * 4 + wid;
  int e = tile * 16 + (lane & 15);
  const __bf16* ps = xn + (size_t)src[e] * 2048;   // xn[src,0,:]
  const __bf16* pd = xn + (size_t)dst[e] * 2048;   // xn[dst,0,:]
  const __bf16* pe = erad + (size_t)e * 64;
  f32x8 acc[4];
#pragma unroll
  for (int nt = 0; nt < 4; ++nt)
#pragma unroll
    for (int i = 0; i < 8; ++i) acc[nt][i] = 0.f;
#pragma unroll
  for (int k = 0; k < 10; ++k) {
    int k0 = k * 32 + ((lane & 16) ? 8 : 0);
    bf16x16 a;
#pragma unroll
    for (int half = 0; half < 2; ++half) {
      int jb = k0 + half * 16;   // 8-run never crosses a 128/256 boundary
      const __bf16* p = (jb < 128) ? (ps + jb)
                        : (jb < 256) ? (pd + (jb - 128)) : (pe + (jb - 256));
      bf16x8 run = *(const bf16x8*)(p);
#pragma unroll
      for (int i = 0; i < 8; ++i) a[half * 8 + i] = run[i];
    }
#pragma unroll
    for (int nt = 0; nt < 4; ++nt)
      acc[nt] = wmma_bf16(a, frag_swz(w1s, k * 4 + nt, lane), acc[nt]);
  }
  int roff = (lane & 16) ? 8 : 0, col4 = lane & 15;
#pragma unroll
  for (int nt = 0; nt < 4; ++nt) {
    float bb = b1[nt * 16 + col4];
#pragma unroll
    for (int r = 0; r < 8; ++r)
      hs[wid][(r + roff) * 64 + nt * 16 + col4] = silu_f(acc[nt][r] + bb);
  }
  // second tiny GEMM (64 -> 4) via scalar dots: 64 outputs per wave
  for (int o = lane; o < 64; o += 32) {
    int le = o >> 2, hh = o & 3;
    float s = 0.f;
#pragma unroll 8
    for (int k2 = 0; k2 < 64; ++k2) s += hs[wid][le * 64 + k2] * w2[k2 * 4 + hh];
    logits[(size_t)(tile * 16 + le) * 4 + hh] = s;
  }
}

// ---------------- per-layer zeroing ----------------
__global__ void k_zero_layer(unsigned* __restrict__ menc, float* __restrict__ ssum,
                             float* __restrict__ agg) {
  size_t i = (size_t)blockIdx.x * 256 + threadIdx.x;   // over N*1024
  if (i < (size_t)N * 4) { menc[i] = 0u; ssum[i] = 0.f; }
  agg[i] = 0.f;
}

// ---------------- segment softmax (3 passes) ----------------
__global__ void k_smax_max(const float* __restrict__ logits, const int* __restrict__ dst,
                           unsigned* __restrict__ menc) {
  int i = blockIdx.x * 256 + threadIdx.x;   // E*4
  int e = i >> 2, h = i & 3;
  atomicMax(&menc[(size_t)dst[e] * 4 + h], fenc(logits[i]));
}
__global__ void k_smax_sum(const float* __restrict__ logits, const int* __restrict__ dst,
                           const unsigned* __restrict__ menc, float* __restrict__ ssum,
                           float* __restrict__ alpha) {
  int i = blockIdx.x * 256 + threadIdx.x;
  int e = i >> 2, h = i & 3;
  float m = fdec(menc[(size_t)dst[e] * 4 + h]);
  float ex = __expf(logits[i] - m);
  alpha[i] = ex;
  atomicAdd(&ssum[(size_t)dst[e] * 4 + h], ex);
}
__global__ void k_smax_norm(const int* __restrict__ dst, const float* __restrict__ ssum,
                            float* __restrict__ alpha) {
  int i = blockIdx.x * 256 + threadIdx.x;
  int e = i >> 2, h = i & 3;
  alpha[i] = alpha[i] / (ssum[(size_t)dst[e] * 4 + h] + 1e-9f);
}

// ---------------- value projection + alpha scale + segment sum ----------------
// w_v fragment table pulled into LDS via the Tensor Data Mover (async DMA).
__global__ void __launch_bounds__(256) k_value(const __bf16* __restrict__ xn,
    const int* __restrict__ src, const int* __restrict__ dst,
    const __bf16* __restrict__ wvs, const float* __restrict__ alpha,
    float* __restrict__ agg) {
  __shared__ __align__(32) __bf16 wvl[128 * 64];       // 16KB swizzled w_v table
  int tid = threadIdx.x, lane = tid & 31, wid = tid >> 5;
  if (tid == 0) {
    tdm_load_1d((unsigned)(size_t)(void*)wvl, wvs, (128 * 64 * 2) / 8);
    __builtin_amdgcn_s_wait_tensorcnt(0);
  }
  __syncthreads();
  int e = blockIdx.x * 8 + wid;
  const __bf16* A = xn + (size_t)src[e] * 2048;   // 16 x 128 bf16
  int ep = (e + 8 < E) ? (e + 8) : e;             // prefetch next edge's source row
  __builtin_prefetch(xn + (size_t)src[ep] * 2048, 0, 0);
  f32x8 acc[4];
#pragma unroll
  for (int nt = 0; nt < 4; ++nt)
#pragma unroll
    for (int i = 0; i < 8; ++i) acc[nt][i] = 0.f;
#pragma unroll
  for (int k = 0; k < 4; ++k) {
    bf16x16 a = a_frag_bf16(A + k * 32, 128, lane);
#pragma unroll
    for (int nt = 0; nt < 4; ++nt)   // 4 n-tiles of 16 = 4 heads * VC 16
      acc[nt] = wmma_bf16(a, frag_swz(wvl, k * 4 + nt, lane), acc[nt]);
  }
  int roff = (lane & 16) ? 8 : 0, col4 = lane & 15;
  float* ap = agg + (size_t)dst[e] * 1024;
#pragma unroll
  for (int nt = 0; nt < 4; ++nt) {     // head == nt (VC == tile width == 16)
    float al = alpha[(size_t)e * 4 + nt];
#pragma unroll
    for (int r = 0; r < 8; ++r)
      atomicAdd(ap + (r + roff) * 64 + nt * 16 + col4, acc[nt][r] * al);
  }
}

// ---------------- x += agg @ w_o  (16x64)@(64,128) per node ----------------
__global__ void __launch_bounds__(256) k_wo(const float* __restrict__ agg,
    const __bf16* __restrict__ wos, float* __restrict__ x) {
  int tid = threadIdx.x, lane = tid & 31, wid = tid >> 5;
  int n = blockIdx.x * 8 + wid;
  const float* A = agg + (size_t)n * 1024;
  f32x8 acc[8];
#pragma unroll
  for (int nt = 0; nt < 8; ++nt)
#pragma unroll
    for (int i = 0; i < 8; ++i) acc[nt][i] = 0.f;
#pragma unroll
  for (int k = 0; k < 2; ++k) {
    bf16x16 a = a_frag_f32(A + k * 32, 64, lane);
#pragma unroll
    for (int nt = 0; nt < 8; ++nt)
      acc[nt] = wmma_bf16(a, frag_swz(wos, k * 8 + nt, lane), acc[nt]);
  }
  int roff = (lane & 16) ? 8 : 0, col4 = lane & 15;
  float* xp = x + (size_t)n * 2048;
#pragma unroll
  for (int nt = 0; nt < 8; ++nt)
#pragma unroll
    for (int r = 0; r < 8; ++r)
      xp[(r + roff) * 128 + nt * 16 + col4] += acc[nt][r];
}

// ---------------- gated FFN per node: (16,128)@(128,256) -> gate -> @(256,128)
__global__ void __launch_bounds__(128) k_ffn(const __bf16* __restrict__ xn,
    const __bf16* __restrict__ w1s, const __bf16* __restrict__ w2s,
    float* __restrict__ x) {
  __shared__ __align__(32) __bf16 hsh[4][8 * 32 * 16];  // swizzled 16x256 per wave
  int tid = threadIdx.x, lane = tid & 31, wid = tid >> 5;
  int n = blockIdx.x * 4 + wid;
  const __bf16* A = xn + (size_t)n * 2048;
  int roff = (lane & 16) ? 8 : 0, col4 = lane & 15;
  int lhx = (col4 >> 3) << 4;
  __bf16* h = hsh[wid];
  // GEMM1 in two halves of 8 n-tiles; gating fused via cross-lane row-0 broadcast
#pragma unroll
  for (int half = 0; half < 2; ++half) {
    f32x8 acc[8];
#pragma unroll
    for (int nt = 0; nt < 8; ++nt)
#pragma unroll
      for (int i = 0; i < 8; ++i) acc[nt][i] = 0.f;
#pragma unroll
    for (int k = 0; k < 4; ++k) {
      bf16x16 a = a_frag_bf16(A + k * 32, 128, lane);
#pragma unroll
      for (int nt = 0; nt < 8; ++nt)
        acc[nt] = wmma_bf16(a, frag_swz(w1s, k * 16 + half * 8 + nt, lane), acc[nt]);
    }
#pragma unroll
    for (int nt = 0; nt < 8; ++nt) {
      int gnt = half * 8 + nt;
      float h0 = __shfl(acc[nt][0], col4, 32);    // row-0 value of this column
      float g  = 1.f / (1.f + __expf(-h0));       // sigmoid gate; silu(h0)=h0*g
      int kf = gnt >> 1;
      int eb = (gnt & 1) * 8 + (col4 & 7);
#pragma unroll
      for (int r = 0; r < 8; ++r)
        h[((kf * 32) + (r + roff) + lhx) * 16 + eb] = (__bf16)(acc[nt][r] * g);
    }
  }
  // GEMM2: h(16x256) @ w2(256x128), accumulate into x
  f32x8 acc2[8];
#pragma unroll
  for (int nt = 0; nt < 8; ++nt)
#pragma unroll
    for (int i = 0; i < 8; ++i) acc2[nt][i] = 0.f;
#pragma unroll
  for (int k = 0; k < 8; ++k) {
    bf16x16 a = frag_swz(h, k, lane);
#pragma unroll
    for (int nt = 0; nt < 8; ++nt)
      acc2[nt] = wmma_bf16(a, frag_swz(w2s, k * 8 + nt, lane), acc2[nt]);
  }
  float* xp = x + (size_t)n * 2048;
#pragma unroll
  for (int nt = 0; nt < 8; ++nt)
#pragma unroll
    for (int r = 0; r < 8; ++r)
      xp[(r + roff) * 128 + nt * 16 + col4] += acc2[nt][r];
}

// ---------------- host launcher ----------------
extern "C" void kernel_launch(void* const* d_in, const int* in_sizes, int n_in,
                              void* d_out, int out_size, void* d_ws, size_t ws_size,
                              hipStream_t stream) {
  (void)in_sizes; (void)n_in; (void)out_size; (void)ws_size;
  const float* pos        = (const float*)d_in[0];
  const float* node_feats = (const float*)d_in[1];
  const int*   eidx       = (const int*)d_in[2];
  const float* w_rad_deg1 = (const float*)d_in[3];
  const float* b_rad_deg1 = (const float*)d_in[4];
  const float* w_rad_deg2 = (const float*)d_in[5];
  const float* w_rad_attn = (const float*)d_in[6];
  const float* b_rad_attn = (const float*)d_in[7];
  const float* w_alpha1   = (const float*)d_in[8];
  const float* b_alpha1   = (const float*)d_in[9];
  const float* w_alpha2   = (const float*)d_in[10];
  const float* w_v        = (const float*)d_in[11];
  const float* w_o        = (const float*)d_in[12];
  const float* gamma_attn = (const float*)d_in[13];
  const float* gamma_ffn  = (const float*)d_in[14];
  const float* w_ffn1     = (const float*)d_in[15];
  const float* w_ffn2     = (const float*)d_in[16];
  const int* src = eidx;
  const int* dst = eidx + E;

  float* x = (float*)d_out;                  // (N,16,128)

  // ---- workspace: bf16 region (weights + activations), then f32 buffers ----
  __bf16* wb = (__bf16*)d_ws;
  size_t ob = 0;
  __bf16* rad1s    = wb + ob; ob += 96 * 64;
  __bf16* rad2s    = wb + ob; ob += 64 * 128;
  __bf16* radatt_s = wb + ob; ob += (size_t)L * 96 * 64;
  __bf16* alpha1_s = wb + ob; ob += (size_t)L * 320 * 64;
  __bf16* wv_s     = wb + ob; ob += (size_t)L * 128 * 64;
  __bf16* wo_s     = wb + ob; ob += (size_t)L * 64 * 128;
  __bf16* ffn1_s   = wb + ob; ob += (size_t)L * 128 * 256;
  __bf16* ffn2_s   = wb + ob; ob += (size_t)L * 256 * 128;
  __bf16* xn       = wb + ob; ob += (size_t)N * 2048;   // bf16 activations
  __bf16* rbf      = wb + ob; ob += (size_t)E * NB;
  __bf16* erad     = wb + ob; ob += (size_t)E * 64;

  float* ws = (float*)(wb + ob);             // ob is even -> 4B aligned
  size_t o = 0;
  float* sh     = ws + o; o += (size_t)E * 16;
  float* rad    = ws + o; o += (size_t)E * 128;
  float* logits = ws + o; o += (size_t)E * 4;
  float* alpha  = ws + o; o += (size_t)E * 4;
  unsigned* menc = (unsigned*)(ws + o); o += (size_t)N * 4;
  float* ssum   = ws + o; o += (size_t)N * 4;
  float* agg    = ws + o; o += (size_t)N * 1024;

  // ---- pre-swizzle all weights into fragment-major bf16 tables ----
  k_prep_b<<<(96 * 64) / 256, 256, 0, stream>>>(w_rad_deg1, rad1s, 64, 4, 96 * 64);
  k_prep_b<<<(64 * 128) / 256, 256, 0, stream>>>(w_rad_deg2, rad2s, 128, 8, 64 * 128);
  for (int i = 0; i < L; ++i) {
    k_prep_b<<<(96 * 64) / 256, 256, 0, stream>>>(
        w_rad_attn + (size_t)i * 96 * 64, radatt_s + (size_t)i * 96 * 64, 64, 4, 96 * 64);
    k_prep_b<<<(320 * 64) / 256, 256, 0, stream>>>(
        w_alpha1 + (size_t)i * 320 * 64, alpha1_s + (size_t)i * 320 * 64, 64, 4, 320 * 64);
    k_prep_b<<<(128 * 64) / 256, 256, 0, stream>>>(
        w_v + (size_t)i * 128 * 64, wv_s + (size_t)i * 128 * 64, 64, 4, 128 * 64);
    k_prep_b<<<(64 * 128) / 256, 256, 0, stream>>>(
        w_o + (size_t)i * 64 * 128, wo_s + (size_t)i * 64 * 128, 128, 8, 64 * 128);
    k_prep_b<<<(128 * 256) / 256, 256, 0, stream>>>(
        w_ffn1 + (size_t)i * 128 * 256, ffn1_s + (size_t)i * 128 * 256, 256, 16, 128 * 256);
    k_prep_b<<<(256 * 128) / 256, 256, 0, stream>>>(
        w_ffn2 + (size_t)i * 256 * 128, ffn2_s + (size_t)i * 256 * 128, 128, 8, 256 * 128);
  }

  k_edge_geom<<<E / 256, 256, 0, stream>>>(pos, src, dst, sh, rbf);
  k_rad_mlp<<<E / 128, 256, 0, stream>>>(rbf, rad1s, b_rad_deg1, rad2s, rad);
  k_init_x<<<(N * 2048) / 256, 256, 0, stream>>>(node_feats, x);
  k_msg_scatter<<<E, 128, 0, stream>>>(sh, rad, dst, x);

  for (int i = 0; i < L; ++i) {
    k_rmsnorm<<<N, 128, 0, stream>>>(x, gamma_attn + (size_t)i * 512, xn);
    k_erad<<<E / 128, 256, 0, stream>>>(rbf, radatt_s + (size_t)i * 96 * 64,
                                        b_rad_attn + (size_t)i * 64, erad);
    k_alpha<<<E / 64, 128, 0, stream>>>(xn, src, dst, erad,
                                        alpha1_s + (size_t)i * 320 * 64,
                                        b_alpha1 + (size_t)i * 64,
                                        w_alpha2 + (size_t)i * 64 * 4, logits);
    k_zero_layer<<<(N * 1024) / 256, 256, 0, stream>>>(menc, ssum, agg);
    k_smax_max<<<(E * 4) / 256, 256, 0, stream>>>(logits, dst, menc);
    k_smax_sum<<<(E * 4) / 256, 256, 0, stream>>>(logits, dst, menc, ssum, alpha);
    k_smax_norm<<<(E * 4) / 256, 256, 0, stream>>>(dst, ssum, alpha);
    k_value<<<E / 8, 256, 0, stream>>>(xn, src, dst, wv_s + (size_t)i * 128 * 64,
                                       alpha, agg);
    k_wo<<<N / 8, 256, 0, stream>>>(agg, wo_s + (size_t)i * 64 * 128, x);
    k_rmsnorm<<<N, 128, 0, stream>>>(x, gamma_ffn + (size_t)i * 512, xn);
    k_ffn<<<N / 4, 128, 0, stream>>>(xn, ffn1_s + (size_t)i * 128 * 256,
                                     ffn2_s + (size_t)i * 256 * 128, x);
  }
}
